// SequenceEncoderMaskedLSTM_37400575213987
// MI455X (gfx1250) — matrix operand order repaired and checked by reading, not compiled, not run on hardware
//
#include <hip/hip_runtime.h>
#include <hip/hip_bf16.h>

// ---------------------------------------------------------------------------
// Masked-LSTM sequence encoder for gfx1250 (CDNA5).
// bf16 WMMA gates GEMM with async-to-LDS double-buffered weight staging,
// f32 cell math, fused next-step embed. 102 stream-ordered launches.
// N = 1024 sequences, S = 100, D = 64, H = 512, gates = 4H = 2048.
// ---------------------------------------------------------------------------

#define N_SEQ 1024
#define S_LEN 100
#define D_IN  64
#define H_DIM 512
#define G_DIM 2048   // 4H
#define K_CAT 1024   // [xt | h]

#define B_ROWS    64          // 4 gates x 16 cols of W rows per block
#define B_PITCH   80          // bytes per LDS row (64B data + 16B pad, 16B aligned)
#define B_BUFSZ   (B_ROWS * B_PITCH)

typedef __attribute__((ext_vector_type(16))) __bf16 v16bf;
typedef __attribute__((ext_vector_type(8)))  float  v8f;

union BF16x16 {
    uint4          q[2];
    unsigned short s[16];
    v16bf          v;
};

__device__ __forceinline__ unsigned short f2bf(float f) {
    unsigned int u = __float_as_uint(f);
    u += 0x7FFFu + ((u >> 16) & 1u);      // round-to-nearest-even
    return (unsigned short)(u >> 16);
}

__device__ __forceinline__ float sigmoidf_(float x) {
    return 1.0f / (1.0f + __expf(-x));
}

// CDNA5 async global->LDS copy: 16B per lane, tracked by ASYNCcnt.
__device__ __forceinline__ void async_load_b128(unsigned lds_addr, const void* gptr) {
    asm volatile("global_load_async_to_lds_b128 %0, %1, off"
                 :: "v"(lds_addr), "v"(gptr) : "memory");
}
__device__ __forceinline__ void wait_asynccnt0() {
    asm volatile("s_wait_asynccnt 0x0" ::: "memory");
}

// ---------------------------------------------------------------------------
// Kernel 1: one-time weight conversion.
//   Wcat_bf = [W_ih | W_hh] (2048 x 1024 bf16), We_bf, bias = b_ih + b_hh
// ---------------------------------------------------------------------------
__global__ void prep_kernel(const float* __restrict__ We,
                            const float* __restrict__ Wih,
                            const float* __restrict__ Whh,
                            const float* __restrict__ bih,
                            const float* __restrict__ bhh,
                            unsigned short* __restrict__ We_bf,
                            unsigned short* __restrict__ Wcat_bf,
                            float* __restrict__ bias) {
    int idx    = blockIdx.x * blockDim.x + threadIdx.x;
    int stride = gridDim.x * blockDim.x;
    const int total = G_DIM * K_CAT;
    for (int i = idx; i < total; i += stride) {
        int j = i >> 10;
        int k = i & (K_CAT - 1);
        float w = (k < H_DIM) ? Wih[j * H_DIM + k] : Whh[j * H_DIM + (k - H_DIM)];
        Wcat_bf[i] = f2bf(w);
    }
    for (int i = idx; i < H_DIM * D_IN; i += stride) We_bf[i] = f2bf(We[i]);
    for (int i = idx; i < G_DIM; i += stride)        bias[i]  = bih[i] + bhh[i];
}

// ---------------------------------------------------------------------------
// Embed helper: one 16x16 tile of e = x[.,t,:] @ We^T  (K = 64, 2 WMMA).
// ---------------------------------------------------------------------------
__device__ __forceinline__ v8f embed_tile(const float* __restrict__ x,
                                          const unsigned short* __restrict__ We_bf,
                                          int rowbase, int t, int l16, int hi, int j) {
    const float* xrow = x + ((size_t)(rowbase + l16) * S_LEN + t) * D_IN;
    const unsigned short* wrow = We_bf + (size_t)j * D_IN;
    v8f e = {0.f, 0.f, 0.f, 0.f, 0.f, 0.f, 0.f, 0.f};
    BF16x16 A, B;
#pragma unroll
    for (int kk = 0; kk < 2; ++kk) {
        const int k0  = kk * 32;
        const int kbA = k0 + (hi ? 8 : 0);
        float4 f0 = *(const float4*)(xrow + kbA);
        float4 f1 = *(const float4*)(xrow + kbA + 4);
        float4 f2 = *(const float4*)(xrow + kbA + 16);
        float4 f3 = *(const float4*)(xrow + kbA + 20);
        A.s[0]  = f2bf(f0.x); A.s[1]  = f2bf(f0.y); A.s[2]  = f2bf(f0.z); A.s[3]  = f2bf(f0.w);
        A.s[4]  = f2bf(f1.x); A.s[5]  = f2bf(f1.y); A.s[6]  = f2bf(f1.z); A.s[7]  = f2bf(f1.w);
        A.s[8]  = f2bf(f2.x); A.s[9]  = f2bf(f2.y); A.s[10] = f2bf(f2.z); A.s[11] = f2bf(f2.w);
        A.s[12] = f2bf(f3.x); A.s[13] = f2bf(f3.y); A.s[14] = f2bf(f3.z); A.s[15] = f2bf(f3.w);
        B.q[0] = *(const uint4*)(wrow + k0 + hi * 8);
        B.q[1] = *(const uint4*)(wrow + k0 + hi * 8 + 16);
        e = __builtin_amdgcn_wmma_f32_16x16x32_bf16(false, A.v, false, B.v,
                                                    (short)0, e, false, false);
    }
    return e;
}

// ---------------------------------------------------------------------------
// Kernel 2: init. One wave per 16x16 tile (2048 waves).
//   h(0) = sel(m0, e0, h0); c = c0; Apanel0 = [ bf16(xt0) | bf16(h(0)) ]
// ---------------------------------------------------------------------------
__global__ void init_kernel(const float* __restrict__ x,
                            const int* __restrict__ mask,
                            const unsigned short* __restrict__ We_bf,
                            const float* __restrict__ be,
                            const float* __restrict__ h0,
                            const float* __restrict__ c0,
                            float* __restrict__ h,
                            float* __restrict__ cst,
                            unsigned short* __restrict__ Aout) {
    const int lane = threadIdx.x & 31;
    const int wave = blockIdx.x * (blockDim.x >> 5) + (threadIdx.x >> 5);
    const int r   = wave >> 5;   // row tile 0..63
    const int cc  = wave & 31;   // col tile 0..31
    const int hi  = lane >> 4;
    const int l16 = lane & 15;
    const int j   = cc * 16 + l16;

    v8f e = embed_tile(x, We_bf, r * 16, 0, l16, hi, j);

    const float bej = be[j];
    const float h0j = h0[j];
    const float c0j = c0[j];
#pragma unroll
    for (int v = 0; v < 8; ++v) {
        const int nrow = r * 16 + v + hi * 8;
        const float ev = e[v] + bej;
        const bool  m0 = mask[(size_t)nrow * S_LEN + 0] != 0;
        const float hv = m0 ? ev : h0j;            // h(0)
        const float xt = m0 ? ev : hv;             // xt for scan step 0
        const size_t o = (size_t)nrow * H_DIM + j;
        h[o]   = hv;
        cst[o] = c0j;
        Aout[(size_t)nrow * K_CAT + j]         = f2bf(xt);
        Aout[(size_t)nrow * K_CAT + H_DIM + j] = f2bf(hv);
    }
}

// ---------------------------------------------------------------------------
// Kernel 3: fused LSTM step.
// Block = 8 waves sharing ONE column tile (ccB): waves cover 8 x 32 rows.
// Per 32-wide K chunk, the block async-copies the 4KB B tile (64 Wcat rows)
// into LDS (double-buffered), each wave ds_loads its B fragments and runs
// 8 WMMAs against its private A rows. Then f32 LSTM cell + fused embed(t+1).
// ---------------------------------------------------------------------------
__global__ void lstm_step_kernel(const float* __restrict__ x,
                                 const int* __restrict__ mask,
                                 const unsigned short* __restrict__ We_bf,
                                 const float* __restrict__ be,
                                 const unsigned short* __restrict__ Wcat,
                                 const float* __restrict__ bias,
                                 const unsigned short* __restrict__ Ain,
                                 unsigned short* __restrict__ Aout,
                                 float* __restrict__ cst,
                                 float* __restrict__ h,
                                 int t) {
    __shared__ __align__(16) unsigned char Bsm[2 * B_BUFSZ];

    const int tid  = threadIdx.x;
    const int lane = tid & 31;
    const int widx = tid >> 5;                // wave in block 0..7
    const int ccB  = blockIdx.x & 31;         // column tile for whole block
    const int rg   = blockIdx.x >> 5;         // row group 0..3
    const int r2   = rg * 8 + widx;           // 32-row block 0..31
    const int hi   = lane >> 4;
    const int l16  = lane & 15;
    const int j    = ccB * 16 + l16;

    // async B staging: thread -> one 16B segment of the 64x64B chunk tile
    const int brow = tid >> 2;                // 0..63  (g = brow>>4, n = brow&15)
    const int bseg = tid & 3;                 // 16B segment in row
    const unsigned short* bsrc = Wcat +
        (size_t)((brow >> 4) * H_DIM + ccB * 16 + (brow & 15)) * K_CAT + bseg * 8;
    const unsigned ldsrow = (unsigned)(brow * B_PITCH + bseg * 16);

    const unsigned short* arow0 = Ain + (size_t)(r2 * 32 + l16) * K_CAT;
    const unsigned short* arow1 = arow0 + (size_t)16 * K_CAT;

    v8f acc[4][2];
#pragma unroll
    for (int g = 0; g < 4; ++g) {
        const float bg = bias[g * H_DIM + j];
#pragma unroll
        for (int s = 0; s < 2; ++s)
#pragma unroll
            for (int v = 0; v < 8; ++v) acc[g][s][v] = bg;
    }

    // prologue: stage chunk 0
    async_load_b128((unsigned)(size_t)&Bsm[ldsrow], bsrc);
    wait_asynccnt0();
    __syncthreads();

    // ---- gates GEMM: K = 1024, 32 chunks, 8 WMMA per chunk ----------------
    BF16x16 A0, A1, Bg;
    for (int kk = 0; kk < 32; ++kk) {
        const int cur = kk & 1;
        if (kk + 1 < 32) {                    // prefetch next chunk into alt buf
            async_load_b128((unsigned)(size_t)&Bsm[((kk + 1) & 1) * B_BUFSZ + ldsrow],
                            bsrc + (size_t)(kk + 1) * 32);
        }
        const int k0  = kk * 32;
        const int kbA = k0 + (hi ? 8 : 0);
        A0.q[0] = *(const uint4*)(arow0 + kbA);
        A0.q[1] = *(const uint4*)(arow0 + kbA + 16);
        A1.q[0] = *(const uint4*)(arow1 + kbA);
        A1.q[1] = *(const uint4*)(arow1 + kbA + 16);
        const unsigned char* bbuf = &Bsm[cur * B_BUFSZ];
#pragma unroll
        for (int g = 0; g < 4; ++g) {
            const unsigned char* brd = bbuf + (g * 16 + l16) * B_PITCH + hi * 16;
            Bg.q[0] = *(const uint4*)brd;
            Bg.q[1] = *(const uint4*)(brd + 32);
            acc[g][0] = __builtin_amdgcn_wmma_f32_16x16x32_bf16(false, A0.v, false, Bg.v,
                                                                (short)0, acc[g][0], false, false);
            acc[g][1] = __builtin_amdgcn_wmma_f32_16x16x32_bf16(false, A1.v, false, Bg.v,
                                                                (short)0, acc[g][1], false, false);
        }
        wait_asynccnt0();                     // own async copies done
        __syncthreads();                      // => all waves' copies done / reads done
    }

    // ---- LSTM cell in f32 registers ---------------------------------------
    float hnew[2][8];
#pragma unroll
    for (int s = 0; s < 2; ++s) {
#pragma unroll
        for (int v = 0; v < 8; ++v) {
            const int nrow = r2 * 32 + s * 16 + v + hi * 8;
            const size_t o = (size_t)nrow * H_DIM + j;
            const float i_ = sigmoidf_(acc[0][s][v]);
            const float f_ = sigmoidf_(acc[1][s][v]);
            const float g_ = tanhf(acc[2][s][v]);
            const float o_ = sigmoidf_(acc[3][s][v]);
            const float cn = f_ * cst[o] + i_ * g_;
            cst[o] = cn;
            const float hv = o_ * tanhf(cn);
            h[o] = hv;
            hnew[s][v] = hv;
        }
    }

    // ---- embed t+1 and build next A-panel (skipped on last step) ----------
    if (t + 1 < S_LEN) {
        const float bej = be[j];
#pragma unroll
        for (int s = 0; s < 2; ++s) {
            const int rowbase = r2 * 32 + s * 16;
            v8f e = embed_tile(x, We_bf, rowbase, t + 1, l16, hi, j);
#pragma unroll
            for (int v = 0; v < 8; ++v) {
                const int nrow = rowbase + v + hi * 8;
                const float ev = e[v] + bej;
                const bool  mv = mask[(size_t)nrow * S_LEN + (t + 1)] != 0;
                const float hv = hnew[s][v];
                const float xt = mv ? ev : hv;   // mask is exactly 0/1 -> select
                Aout[(size_t)nrow * K_CAT + j]         = f2bf(xt);
                Aout[(size_t)nrow * K_CAT + H_DIM + j] = f2bf(hv);
            }
        }
    }
}

// ---------------------------------------------------------------------------
// Host-side launch: prep, init, then 100 fused stream-ordered LSTM steps.
// ---------------------------------------------------------------------------
extern "C" void kernel_launch(void* const* d_in, const int* in_sizes, int n_in,
                              void* d_out, int out_size, void* d_ws, size_t ws_size,
                              hipStream_t stream) {
    const float* x    = (const float*)d_in[0];   // (16,64,100,64) f32
    const int*   mask = (const int*)d_in[1];     // (16,64,100)
    const float* We   = (const float*)d_in[2];   // (512,64)
    const float* be   = (const float*)d_in[3];   // (512,)
    const float* Wih  = (const float*)d_in[4];   // (2048,512)
    const float* Whh  = (const float*)d_in[5];   // (2048,512)
    const float* bih  = (const float*)d_in[6];   // (2048,)
    const float* bhh  = (const float*)d_in[7];   // (2048,)
    const float* h0   = (const float*)d_in[8];   // (1,512)
    const float* c0   = (const float*)d_in[9];   // (1,512)
    float* h = (float*)d_out;                    // (1024,512) f32 = h state

    // workspace carve-up (~10.6 MB)
    char* ws = (char*)d_ws;
    unsigned short* Wcat_bf = (unsigned short*)ws; ws += (size_t)G_DIM * K_CAT * 2;  // 4 MB
    unsigned short* We_bf   = (unsigned short*)ws; ws += (size_t)H_DIM * D_IN * 2;   // 64 KB
    float*          bias    = (float*)ws;          ws += (size_t)G_DIM * 4;          // 8 KB
    float*          cst     = (float*)ws;          ws += (size_t)N_SEQ * H_DIM * 4;  // 2 MB
    unsigned short* Apan0   = (unsigned short*)ws; ws += (size_t)N_SEQ * K_CAT * 2;  // 2 MB
    unsigned short* Apan1   = (unsigned short*)ws;                                   // 2 MB

    prep_kernel<<<1024, 256, 0, stream>>>(We, Wih, Whh, bih, bhh, We_bf, Wcat_bf, bias);

    // init: 2048 wave-tiles, 8 waves (256 thr) per block -> 256 blocks
    init_kernel<<<256, 256, 0, stream>>>(x, mask, We_bf, be, h0, c0, h, cst, Apan0);

    // fused steps: 128 blocks x 256 threads (8 waves sharing a column tile)
    for (int t = 0; t < S_LEN; ++t) {
        unsigned short* Ain  = (t & 1) ? Apan1 : Apan0;
        unsigned short* Aout = (t & 1) ? Apan0 : Apan1;
        lstm_step_kernel<<<128, 256, 0, stream>>>(x, mask, We_bf, be, Wcat_bf, bias,
                                                  Ain, Aout, cst, h, t);
    }
}
